// DefEDNet_66735201845391
// MI455X (gfx1250) — matrix operand order repaired and checked by reading, not compiled, not run on hardware
//
#include <hip/hip_runtime.h>
#include <hip/hip_bf16.h>

typedef __attribute__((ext_vector_type(16))) _Float16 v16h;
typedef __attribute__((ext_vector_type(8)))  float    v8f;
typedef __attribute__((ext_vector_type(4)))  int      v4i;

#define AS_GLOBAL __attribute__((address_space(1)))
#define AS_LOCAL  __attribute__((address_space(3)))

namespace {
constexpr int kB = 4;
constexpr int kC = 64;
constexpr int kH = 128;
constexpr int kW = 128;
constexpr int kN = 9;     // 3x3 taps
constexpr int kOffC = 18; // 2*kN
constexpr int kOutC = 64;
constexpr int kSpatial = kB * kH * kW; // 65536
}

#if __has_builtin(__builtin_amdgcn_global_load_async_to_lds_b128) && \
    __has_builtin(__builtin_amdgcn_s_wait_asynccnt)
#define USE_ASYNC_LDS 1
#else
#define USE_ASYNC_LDS 0
#endif

// ---------------------------------------------------------------------------
// Kernel 1: depthwise 3x3 (pad 1) on x, plus transpose both the dw result and
// x itself into f16 channel-last layout [(b,h,w), c] via LDS.
// One block per (b, h) row. 256 threads.
// ---------------------------------------------------------------------------
__global__ __launch_bounds__(256) void k_dw1(const float* __restrict__ x,
                                             const float* __restrict__ p_dw,
                                             _Float16* __restrict__ t,
                                             _Float16* __restrict__ xT) {
  __shared__ _Float16 lt[kC][kW + 2]; // +2 pad: avoid bank conflicts on transpose
  __shared__ _Float16 lx[kC][kW + 2];

  const int b = blockIdx.x / kH;
  const int h = blockIdx.x % kH;
  const float* xb = x + (size_t)b * kC * kH * kW;

  for (int idx = threadIdx.x; idx < kC * kW; idx += 256) {
    const int c = idx / kW;  // w fastest -> coalesced x reads
    const int w = idx % kW;
    const float* xc = xb + (size_t)c * kH * kW;
    const float* wk = p_dw + c * 9;
    float acc = 0.f;
#pragma unroll
    for (int i = 0; i < 3; ++i) {
      const int hh = h - 1 + i;
      if (hh < 0 || hh >= kH) continue;
#pragma unroll
      for (int j = 0; j < 3; ++j) {
        const int ww = w - 1 + j;
        if (ww < 0 || ww >= kW) continue;
        acc += wk[i * 3 + j] * xc[hh * kW + ww];
      }
    }
    lt[c][w] = (_Float16)acc;
    lx[c][w] = (_Float16)xc[h * kW + w];
  }
  __syncthreads();

  const size_t rowb = (size_t)(b * kH + h) * kW;
  for (int idx = threadIdx.x; idx < kC * kW; idx += 256) {
    const int c = idx % kC;  // c fastest -> coalesced channel-last writes
    const int w = idx / kC;
    t [(rowb + w) * kC + c] = lt[c][w];
    xT[(rowb + w) * kC + c] = lx[c][w];
  }
}

// ---------------------------------------------------------------------------
// Weight pre-pack: build per-lane WMMA B fragments, f16, zero-padded to
// NT*16 output channels. Layout: Bp[((nt*2+kc)*32 + lane)*16 + u] where
//   o = nt*16 + (lane&15), k = kc*32 + (lane<16 ? 0 : 8) + (u&7) + (u>>3)*16
// (matches the ISA 16-bit B 32x16 layout: lanes 0-15 hold K 0..7,16..23 of the
//  chunk, lanes 16-31 hold K 8..15,24..31; one column per lane).
// ---------------------------------------------------------------------------
template <int OC>
__global__ __launch_bounds__(256) void k_pack_w(const float* __restrict__ Wt,
                                                _Float16* __restrict__ Bp) {
  constexpr int NT = (OC + 15) / 16;
  constexpr int NEL = NT * 2 * 32 * 16;
  const int i = (int)(blockIdx.x * blockDim.x + threadIdx.x);
  if (i >= NEL) return;
  const int u    = i & 15;
  const int lane = (i >> 4) & 31;
  const int kc   = (i >> 9) & 1;
  const int nt   = i >> 10;
  const int o = nt * 16 + (lane & 15);
  const int k = kc * 32 + ((lane < 16) ? 0 : 8) + (u & 7) + ((u >> 3) << 4);
  Bp[i] = (o < OC) ? (_Float16)Wt[o * kC + k] : (_Float16)0.0f;
}

// ---------------------------------------------------------------------------
// WMMA pointwise conv (1x1) = GEMM: out[o, s] = sum_k Wt[o,k] * A[s,k]
// A: (kSpatial, 64) f16 channel-last. Bp: pre-packed fragments (see above).
// CHLAST=true  -> out[s*OC + o]  (f32 channel-last, used for the offsets)
// CHLAST=false -> out[(b,OC,h,w)] planar f32 (final output)
// One wave = 16 spatial rows; K=64 as two K=32 WMMA steps; N tiled by 16.
// A tile staged through LDS with gfx1250 async global->LDS copies.
// ---------------------------------------------------------------------------
template <int OC, bool CHLAST>
__global__ __launch_bounds__(256) void k_pw_wmma(const _Float16* __restrict__ A,
                                                 const _Float16* __restrict__ Bp,
                                                 float* __restrict__ out) {
  constexpr int NT = (OC + 15) / 16;
  __shared__ _Float16 atile[8][16 * kC]; // 8 waves x 2KB
  const int warp = (int)(threadIdx.x >> 5);
  const int lane = (int)(threadIdx.x & 31);
  const int wave = (int)blockIdx.x * 8 + warp;
  const int base = wave * 16; // first spatial row of this tile
  _Float16* lt = &atile[warp][0];

  const int M  = lane & 15;
  const int kb = (lane < 16) ? 0 : 8;
  v16h afrag[2];

#if USE_ASYNC_LDS
  // Async-copy this wave's 16x64 f16 A tile (2KB) into LDS: 4 issues of
  // 16B/lane, tracked by ASYNCcnt.
#pragma unroll
  for (int it = 0; it < 4; ++it) {
    const _Float16* g = A + (size_t)base * kC + it * 256 + lane * 8;
    __builtin_amdgcn_global_load_async_to_lds_b128(
        (AS_GLOBAL v4i*)g,
        (AS_LOCAL v4i*)(lt + it * 256 + lane * 8),
        0, 0);
  }
  __builtin_amdgcn_s_wait_asynccnt(0);
  // Fragment gather from LDS (ds_load_b128s: 16B-aligned 8-half runs).
#pragma unroll
  for (int kc = 0; kc < 2; ++kc) {
    const _Float16* r = lt + M * kC + kc * 32;
#pragma unroll
    for (int u = 0; u < 8; ++u) {
      afrag[kc][u]     = r[kb + u];
      afrag[kc][u + 8] = r[16 + kb + u];
    }
  }
#else
  // Fallback: direct global fragment loads.
  const _Float16* arow = A + (size_t)(base + M) * kC;
#pragma unroll
  for (int kc = 0; kc < 2; ++kc) {
#pragma unroll
    for (int u = 0; u < 8; ++u) {
      afrag[kc][u]     = arow[kc * 32 + kb + u];
      afrag[kc][u + 8] = arow[kc * 32 + 16 + kb + u];
    }
  }
#endif

  const v16h* bfr = (const v16h*)Bp; // 32B/lane contiguous fragment loads
#pragma unroll
  for (int nt = 0; nt < NT; ++nt) {
    const int o = nt * 16 + (lane & 15);
    v8f acc = {};
#pragma unroll
    for (int kc = 0; kc < 2; ++kc) {
      const v16h bfrag = bfr[(nt * 2 + kc) * 32 + lane];
      acc = __builtin_amdgcn_wmma_f32_16x16x32_f16(
          /*neg_a=*/false, afrag[kc], /*neg_b=*/false, bfrag,
          /*c_mod=*/(short)0, acc, /*reuse_a=*/false, /*reuse_b=*/false);
    }
    // C/D layout: lane<16 -> M=j, lane>=16 -> M=8+j; N = lane&15
    if (o < OC) {
#pragma unroll
      for (int j = 0; j < 8; ++j) {
        const int m = j + ((lane >> 4) << 3);
        const int s = base + m;
        if (CHLAST) {
          out[(size_t)s * OC + o] = acc[j];
        } else {
          const int w = s & (kW - 1);
          const int h = (s >> 7) & (kH - 1);
          const int b = s >> 14;
          out[(((size_t)b * OC + o) * kH + h) * kW + w] = acc[j];
        }
      }
    }
  }
}

// ---------------------------------------------------------------------------
// Kernel 3: deformable bilinear sampling fused with the stride-3 depthwise
// reduction (y[b,c,h,w] = sum_n c_dw[c,n]*x_off[...,n]).
// One wave per output pixel; each lane owns 2 channels. The 18 offsets of the
// pixel are read with one coalesced load (channel-last) + shuffle broadcasts.
// ---------------------------------------------------------------------------
__global__ __launch_bounds__(256) void k_sample(const _Float16* __restrict__ xT,
                                                const float* __restrict__ offs,
                                                const float* __restrict__ c_dw,
                                                _Float16* __restrict__ y) {
  const int s    = (int)((blockIdx.x * blockDim.x + threadIdx.x) >> 5);
  const int lane = (int)(threadIdx.x & 31);
  const int w = s & (kW - 1);
  const int h = (s >> 7) & (kH - 1);
  const int b = s >> 14;
  const int c0 = lane * 2;

  float wd0[kN], wd1[kN];
#pragma unroll
  for (int n = 0; n < kN; ++n) {
    wd0[n] = c_dw[c0 * kN + n];
    wd1[n] = c_dw[(c0 + 1) * kN + n];
  }

  __builtin_prefetch(offs + (size_t)s * kOffC, 0, 1); // -> global_prefetch_b8
  float myoff = 0.f;
  if (lane < kOffC) myoff = offs[(size_t)s * kOffC + lane];

  const float Hm = (float)(kH + 1); // 129 = Hp-1
  const float Wm = (float)(kW + 1);
  float acc0 = 0.f, acc1 = 0.f;

#pragma unroll
  for (int n = 0; n < kN; ++n) {
    const float px = (float)(h + (n / 3)) + __shfl(myoff, n, 32);
    const float py = (float)(w + (n % 3)) + __shfl(myoff, kN + n, 32);
    const float fx = floorf(px), fy = floorf(py);
    const float ltx = fminf(fmaxf(fx, 0.f), Hm);
    const float lty = fminf(fmaxf(fy, 0.f), Wm);
    const float rbx = fminf(fmaxf(fx + 1.f, 0.f), Hm);
    const float rby = fminf(fmaxf(fy + 1.f, 0.f), Wm);
    const float pxc = fminf(fmaxf(px, 0.f), Hm);
    const float pyc = fminf(fmaxf(py, 0.f), Wm);
    const float g_lt = (1.f + (ltx - pxc)) * (1.f + (lty - pyc));
    const float g_rb = (1.f - (rbx - pxc)) * (1.f - (rby - pyc));
    const float g_lb = (1.f + (ltx - pxc)) * (1.f - (rby - pyc));
    const float g_rt = (1.f - (rbx - pxc)) * (1.f + (lty - pyc));

    float s0 = 0.f, s1 = 0.f;
    auto samp = [&](float qxf, float qyf, float g) {
      const int qx = (int)qxf, qy = (int)qyf; // padded coords in [0, 129]
      if (qx >= 1 && qx <= kH && qy >= 1 && qy <= kW) {
        const _Float16* p =
            xT + (((size_t)(b * kH + (qx - 1))) * kW + (qy - 1)) * kC + c0;
        s0 += g * (float)p[0];
        s1 += g * (float)p[1];
      }
    };
    samp(ltx, lty, g_lt);
    samp(rbx, rby, g_rb);
    samp(ltx, rby, g_lb);
    samp(rbx, lty, g_rt);
    acc0 += wd0[n] * s0;
    acc1 += wd1[n] * s1;
  }

  y[(size_t)s * kC + c0]     = (_Float16)acc0;
  y[(size_t)s * kC + c0 + 1] = (_Float16)acc1;
}

// ---------------------------------------------------------------------------
// Launch. Workspace layout:
//   t    : f16, kSpatial*64      (8 MB)    dw1 output, channel-last
//   xT   : f16, kSpatial*64      (8 MB)    x copy, channel-last
//   offs : f32, kSpatial*18      (4.5 MB)  offset field, channel-last
//   y    : f16, kSpatial*64      (8 MB)    sampled+reduced, channel-last
//   bp1  : f16, 2*2*32*16        (4 KB)    packed B fragments (18x64 pw)
//   bp2  : f16, 4*2*32*16        (8 KB)    packed B fragments (64x64 pw)
// ---------------------------------------------------------------------------
extern "C" void kernel_launch(void* const* d_in, const int* in_sizes, int n_in,
                              void* d_out, int out_size, void* d_ws, size_t ws_size,
                              hipStream_t stream) {
  const float* x    = (const float*)d_in[0];
  const float* p_dw = (const float*)d_in[1];
  const float* p_pw = (const float*)d_in[2];
  const float* c_dw = (const float*)d_in[3];
  const float* c_pw = (const float*)d_in[4];
  float* out = (float*)d_out;

  _Float16* t    = (_Float16*)d_ws;
  _Float16* xT   = t + (size_t)kSpatial * kC;
  float*    offs = (float*)(xT + (size_t)kSpatial * kC);
  _Float16* y    = (_Float16*)(offs + (size_t)kSpatial * kOffC);
  _Float16* bp1  = y + (size_t)kSpatial * kC;
  _Float16* bp2  = bp1 + 2 * 2 * 32 * 16;

  // 0) pre-pack both weight matrices into WMMA B-fragment layout
  k_pack_w<kOffC><<<(2 * 2 * 32 * 16 + 255) / 256, 256, 0, stream>>>(p_pw, bp1);
  k_pack_w<kOutC><<<(4 * 2 * 32 * 16 + 255) / 256, 256, 0, stream>>>(c_pw, bp2);
  // 1) depthwise 3x3 + f16 channel-last transposes
  k_dw1<<<kB * kH, 256, 0, stream>>>(x, p_dw, t, xT);
  // 2) offsets: GEMM 65536 x 18 x 64 via v_wmma_f32_16x16x32_f16
  k_pw_wmma<kOffC, true><<<kSpatial / 16 / 8, 256, 0, stream>>>(t, bp1, offs);
  // 3) deformable sampling + fused stride-3 depthwise reduction
  k_sample<<<kSpatial / 8, 256, 0, stream>>>(xT, offs, c_dw, y);
  // 4) output: GEMM 65536 x 64 x 64 via v_wmma_f32_16x16x32_f16
  k_pw_wmma<kOutC, false><<<kSpatial / 16 / 8, 256, 0, stream>>>(y, bp2, out);
}
